// AugmentedAttentionHead_85607288144433
// MI455X (gfx1250) — matrix operand order repaired and checked by reading, not compiled, not run on hardware
//
#include <hip/hip_runtime.h>
#include <hip/hip_bf16.h>

// ---------------------------------------------------------------------------
// AugmentedAttentionHead for MI455X (gfx1250, wave32, WMMA bf16 + TDM)
// B=8, T=2048, D=768
// ---------------------------------------------------------------------------

#define DD 768
#define TT 2048
#define BTOK 16384   // B*T
#define NBATCH 8

typedef __attribute__((ext_vector_type(16))) __bf16 v16bf;
typedef __attribute__((ext_vector_type(8)))  float  v8f;
typedef __attribute__((ext_vector_type(4)))  unsigned int v4u;
typedef __attribute__((ext_vector_type(8)))  int v8i;
typedef __attribute__((ext_vector_type(4)))  int v4i;

#if __has_builtin(__builtin_amdgcn_tensor_load_to_lds) && \
    __has_builtin(__builtin_amdgcn_s_wait_tensorcnt)
#define USE_TDM 1
#else
#define USE_TDM 0
#endif

union FragU { v16bf v; uint4 u[2]; };

static __device__ __forceinline__ v16bf ld_frag(const void* p0, const void* p1) {
  FragU f;
  f.u[0] = *(const uint4*)p0;
  f.u[1] = *(const uint4*)p1;
  return f.v;
}

static __device__ __forceinline__ unsigned short bf_bits(float f) {
  union { __bf16 h; unsigned short s; } c; c.h = (__bf16)f; return c.s;
}
static __device__ __forceinline__ unsigned pack2(float a, float b) {
  return (unsigned)bf_bits(a) | ((unsigned)bf_bits(b) << 16);
}

#define WMMA_BF16(A, Bm, C) \
  __builtin_amdgcn_wmma_f32_16x16x32_bf16(false, (A), false, (Bm), (short)0, (C), false, false)

// ---------------------------------------------------------------------------
// GEMM tile staging: A [128 x 32] f32->bf16 row-major; W tile transposed into
// sB[n][k] with f32->bf16 conversion.
// ---------------------------------------------------------------------------
static __device__ __forceinline__ void stage_tiles(
    const float* __restrict__ A, const float* __restrict__ W,
    int M0, int N0, int kk, int tid, __bf16* sA, __bf16* sB)
{
  {
    const int m  = tid >> 1;
    const int kh = (tid & 1) * 16;
    const float4* g = (const float4*)(A + (size_t)(M0 + m) * DD + kk + kh);
    float4 f0 = g[0], f1 = g[1], f2 = g[2], f3 = g[3];
    uint4* dst = (uint4*)&sA[m * 32 + kh];
    dst[0] = make_uint4(pack2(f0.x, f0.y), pack2(f0.z, f0.w),
                        pack2(f1.x, f1.y), pack2(f1.z, f1.w));
    dst[1] = make_uint4(pack2(f2.x, f2.y), pack2(f2.z, f2.w),
                        pack2(f3.x, f3.y), pack2(f3.z, f3.w));
  }
  {
    const int kr  = tid >> 3;          // 0..31
    const int seg = (tid & 7) * 16;    // 0..112
    const float4* g = (const float4*)(W + (size_t)(kk + kr) * DD + N0 + seg);
    float4 f0 = g[0], f1 = g[1], f2 = g[2], f3 = g[3];
    float vals[16] = { f0.x, f0.y, f0.z, f0.w, f1.x, f1.y, f1.z, f1.w,
                       f2.x, f2.y, f2.z, f2.w, f3.x, f3.y, f3.z, f3.w };
#pragma unroll
    for (int i = 0; i < 16; i++)
      sB[(seg + i) * 32 + kr] = (__bf16)vals[i];
  }
}

// ---------------------------------------------------------------------------
// Kernel 1: C[16384,768] = x[16384,768] @ W[768,768], bf16 WMMA, f32 accum.
// MODE 0: write f32 row-major (pre-LN q/k).  MODE 1: write bf16 V^T
// (vt[b][d][t]) so the attention PV B-operand is contiguous per lane.
// Block 256 (8 waves). Tile 128x128. Wave (wr,wc): 32x64 sub-tile.
// Double-buffered LDS so global staging overlaps the WMMA chain.
// ---------------------------------------------------------------------------
template <int MODE>
__global__ __launch_bounds__(256)
void gemm768_bf16(const float* __restrict__ A, const float* __restrict__ W,
                  void* __restrict__ outp)
{
  __shared__ __align__(16) __bf16 sA[2][128 * 32];   // [m][k]
  __shared__ __align__(16) __bf16 sB[2][128 * 32];   // [n][k]  (W transposed)

  const int tid  = threadIdx.x;
  const int lane = tid & 31;
  const int w    = tid >> 5;
  const int ln16 = lane & 15;
  const int lh   = lane >> 4;          // which 16-lane half
  const int wr   = w & 3;              // M sub-tile (0..3) * 32
  const int wc   = w >> 2;             // N sub-tile (0..1) * 64
  const int M0   = blockIdx.x * 128;
  const int N0   = blockIdx.y * 128;

  v8f acc[2][4];
#pragma unroll
  for (int mt = 0; mt < 2; mt++)
#pragma unroll
    for (int nt = 0; nt < 4; nt++)
#pragma unroll
      for (int r = 0; r < 8; r++) acc[mt][nt][r] = 0.0f;

  stage_tiles(A, W, M0, N0, 0, tid, sA[0], sB[0]);
  __syncthreads();

  for (int kt = 0; kt < DD / 32; kt++) {
    const int cur = kt & 1;
    if (kt + 1 < DD / 32)
      stage_tiles(A, W, M0, N0, (kt + 1) * 32, tid, sA[cur ^ 1], sB[cur ^ 1]);

    // ---- A fragments (ISA 16-bit A layout: K {0..7,16..23} / {8..15,24..31}) ----
    v16bf af[2];
#pragma unroll
    for (int mt = 0; mt < 2; mt++) {
      const __bf16* p = &sA[cur][(wr * 32 + mt * 16 + ln16) * 32 + (lh ? 8 : 0)];
      af[mt] = ld_frag(p, p + 16);
    }
#pragma unroll
    for (int nt = 0; nt < 4; nt++) {
      const __bf16* p = &sB[cur][(wc * 64 + nt * 16 + ln16) * 32 + lh * 8];
      v16bf bf = ld_frag(p, p + 16);
#pragma unroll
      for (int mt = 0; mt < 2; mt++)
        acc[mt][nt] = WMMA_BF16(af[mt], bf, acc[mt][nt]);
    }
    __syncthreads();
  }

  // ---- epilogue ----
#pragma unroll
  for (int mt = 0; mt < 2; mt++) {
#pragma unroll
    for (int nt = 0; nt < 4; nt++) {
#pragma unroll
      for (int r = 0; r < 8; r++) {
        const int m = M0 + wr * 32 + mt * 16 + r + 8 * lh;
        const int n = N0 + wc * 64 + nt * 16 + ln16;
        const float val = acc[mt][nt][r];
        if (MODE == 0) {
          ((float*)outp)[(size_t)m * DD + n] = val;
        } else {
          const int bb = m >> 11;       // m / 2048
          const int t  = m & 2047;
          ((__bf16*)outp)[((size_t)bb * DD + n) * TT + t] = (__bf16)val;
        }
      }
    }
  }
}

// ---------------------------------------------------------------------------
// Kernel 2: per-row LayerNorm (f32 in -> bf16 out) + optional gates:
//   alpha = softplus(y . Walp + balp),  sigma = sigmoid(y . Wsig[:,0] + bsig[0])
// One 256-thread block per row (768 = 3 * 256 elements per thread).
// ---------------------------------------------------------------------------
__global__ __launch_bounds__(256)
void ln_gate(const float* __restrict__ raw, const float* __restrict__ g,
             const float* __restrict__ beta, __bf16* __restrict__ outb,
             const float* __restrict__ Walp, const float* __restrict__ balp,
             const float* __restrict__ Wsig, const float* __restrict__ bsig,
             float* __restrict__ alpha, float* __restrict__ sigma, int do_gate)
{
  const int row = blockIdx.x;
  const int tid = threadIdx.x;
  const int lane = tid & 31, wid = tid >> 5;
  const float* xr = raw + (size_t)row * DD;

  const float x0 = xr[tid], x1 = xr[tid + 256], x2 = xr[tid + 512];
  float s  = x0 + x1 + x2;
  float s2 = x0 * x0 + x1 * x1 + x2 * x2;

  __shared__ float r1[8], r2[8];
  __shared__ float stats[2];

#pragma unroll
  for (int m = 16; m >= 1; m >>= 1) {
    s  += __shfl_xor(s,  m, 32);
    s2 += __shfl_xor(s2, m, 32);
  }
  if (lane == 0) { r1[wid] = s; r2[wid] = s2; }
  __syncthreads();
  if (tid == 0) {
    float a = 0.f, b = 0.f;
#pragma unroll
    for (int i = 0; i < 8; i++) { a += r1[i]; b += r2[i]; }
    const float mean = a * (1.0f / DD);
    const float var  = b * (1.0f / DD) - mean * mean;
    stats[0] = mean;
    stats[1] = rsqrtf(var + 1e-5f);
  }
  __syncthreads();
  const float mean = stats[0], rstd = stats[1];

  const float y0 = (x0 - mean) * rstd * g[tid]       + beta[tid];
  const float y1 = (x1 - mean) * rstd * g[tid + 256] + beta[tid + 256];
  const float y2 = (x2 - mean) * rstd * g[tid + 512] + beta[tid + 512];

  __bf16* orow = outb + (size_t)row * DD;
  orow[tid]       = (__bf16)y0;
  orow[tid + 256] = (__bf16)y1;
  orow[tid + 512] = (__bf16)y2;

  if (do_gate) {
    float da = y0 * Walp[tid] + y1 * Walp[tid + 256] + y2 * Walp[tid + 512];
    float ds = y0 * Wsig[2 * tid] + y1 * Wsig[2 * (tid + 256)] + y2 * Wsig[2 * (tid + 512)];
#pragma unroll
    for (int m = 16; m >= 1; m >>= 1) {
      da += __shfl_xor(da, m, 32);
      ds += __shfl_xor(ds, m, 32);
    }
    __syncthreads();                 // r1/r2 reuse
    if (lane == 0) { r1[wid] = da; r2[wid] = ds; }
    __syncthreads();
    if (tid == 0) {
      float a = 0.f, b = 0.f;
#pragma unroll
      for (int i = 0; i < 8; i++) { a += r1[i]; b += r2[i]; }
      const float za = a + balp[0];
      const float zs = b + bsig[0];
      const float sp = (za > 0.f) ? za + log1pf(__expf(-za)) : log1pf(__expf(za));
      alpha[row] = sp;
      sigma[row] = 1.0f / (1.0f + __expf(-zs));
    }
  }
}

// ---------------------------------------------------------------------------
// Kernel 3: flash attention with post-softmax gates.
//   out[q,:] = (alpha[q]/l[q]) * sum_k exp(s[q,k]-m[q]) * sigma[k] * V[k,:]
//   l[q] = sum_k exp(s[q,k]-m[q])   (ungated denominator)
// Block = 128 threads (4 waves). One WG = (batch, 16-query tile).
// Q tile is brought in by the Tensor Data Mover when available.
// ---------------------------------------------------------------------------
__global__ __launch_bounds__(128)
void attn_flash(const __bf16* __restrict__ q_ln, const __bf16* __restrict__ k_ln,
                const __bf16* __restrict__ vt, const float* __restrict__ alpha,
                const float* __restrict__ sigma, float* __restrict__ out)
{
  __shared__ __align__(16) __bf16 sQ[16 * DD];   // 24 KB, query tile bf16
  __shared__ __align__(16) __bf16 sP[16 * 64];   // 2 KB, gated probs bf16
  __shared__ float sRedM[64];                    // per-wave row maxima
  __shared__ float sRedS[64];                    // per-wave row sums
  __shared__ float sM[16];                       // combined row max
  __shared__ float sL[16];                       // combined row sum

  const int tid  = threadIdx.x;
  const int lane = tid & 31;
  const int w    = tid >> 5;           // 0..3
  const int ln16 = lane & 15;
  const int lh   = lane >> 4;
  const int q0   = blockIdx.x * 16;
  const int bb   = blockIdx.y;

  // ---- load Q tile [16 x 768] bf16 into LDS ----
#if USE_TDM
  if (w == 0) {
    // Tensor DMA descriptor (ISA ch8): 2D tile, 16 rows x 768 bf16.
    const unsigned lds_off = (unsigned)(size_t)(&sQ[0]);
    const unsigned long long ga =
        (unsigned long long)(size_t)(q_ln + (size_t)(bb * TT + q0) * DD);
    v4u g0;
    g0[0] = 1u;                                   // count=1, user descriptor
    g0[1] = lds_off;                              // lds_addr
    g0[2] = (unsigned)ga;                         // global_addr[31:0]
    g0[3] = (unsigned)(ga >> 32) | 0x80000000u;   // global_addr[56:32] | type=2
    v8i g1;
    g1[0] = 0x00010000;            // workgroup_mask=0, data_size=1 (2 bytes)
    g1[1] = (int)(768u << 16);     // tensor_dim0[15:0]  (bits 63:48)
    g1[2] = (int)(16u  << 16);     // tensor_dim0 hi=0, tensor_dim1=16
    g1[3] = (int)(768u << 16);     // tensor_dim1 hi=0, tile_dim0=768
    g1[4] = 16;                    // tile_dim1=16, tile_dim2=0
    g1[5] = 768;                   // tensor_dim0_stride[31:0]
    g1[6] = 0;                     // stride hi, tensor_dim1_stride lo
    g1[7] = 0;
    v4i gz4 = {0, 0, 0, 0};
    v8i gz8 = {0, 0, 0, 0, 0, 0, 0, 0};
    __builtin_amdgcn_tensor_load_to_lds(g0, g1, gz4, gz4, gz8, 0);
    __builtin_amdgcn_s_wait_tensorcnt(0);
  }
#else
  {
    const int row = tid >> 3;
    const int seg = (tid & 7) * 96;
    const uint4* src = (const uint4*)(q_ln + (size_t)(bb * TT + q0 + row) * DD + seg);
    uint4* dst = (uint4*)&sQ[row * DD + seg];
#pragma unroll
    for (int i = 0; i < 12; i++) dst[i] = src[i];
  }
#endif
  __syncthreads();

  v8f o[12];
#pragma unroll
  for (int nt = 0; nt < 12; nt++)
#pragma unroll
    for (int r = 0; r < 8; r++) o[nt][r] = 0.0f;

  float mrow[8], lrow[8];
#pragma unroll
  for (int r = 0; r < 8; r++) { mrow[r] = -1e30f; lrow[r] = 0.0f; }

  const float scl = 0.03608439182435161f;   // 1/sqrt(768)

  for (int kt = 0; kt < TT / 64; kt++) {
    const int kbase = kt * 64;
    const int key   = kbase + w * 16 + ln16;   // this wave's key (per lane)

    // ---- S = Q K^T : one 16x16 tile per wave, K-depth 768 ----
    v8f s;
#pragma unroll
    for (int r = 0; r < 8; r++) s[r] = 0.0f;
    const __bf16* kb = k_ln + (size_t)(bb * TT + key) * DD + lh * 8;
    for (int ks = 0; ks < 24; ks++) {
      const __bf16* ap = &sQ[ln16 * DD + ks * 32 + (lh ? 8 : 0)];
      v16bf a = ld_frag(ap, ap + 16);
      v16bf b = ld_frag(kb + ks * 32, kb + ks * 32 + 16);
      s = WMMA_BF16(a, b, s);
    }
    if (kt + 1 < TT / 64)
      __builtin_prefetch(k_ln + (size_t)(bb * TT + key + 64) * DD, 0, 1);

#pragma unroll
    for (int r = 0; r < 8; r++) s[r] *= scl;

    // ---- per-wave row max, then combine across waves ----
    float tmax[8];
#pragma unroll
    for (int r = 0; r < 8; r++) {
      float t = s[r];
#pragma unroll
      for (int m = 1; m < 16; m <<= 1) t = fmaxf(t, __shfl_xor(t, m, 32));
      tmax[r] = t;
    }
    if (ln16 == 0) {
#pragma unroll
      for (int r = 0; r < 8; r++) sRedM[w * 16 + r + 8 * lh] = tmax[r];
    }
    __syncthreads();
    if (tid < 16) {
      float mm = fmaxf(fmaxf(sRedM[tid], sRedM[16 + tid]),
                       fmaxf(sRedM[32 + tid], sRedM[48 + tid]));
      sM[tid] = mm;
    }
    __syncthreads();

    // ---- online softmax update + gated P tile ----
    const float sg = sigma[(size_t)bb * TT + key];
    float corr[8];
#pragma unroll
    for (int r = 0; r < 8; r++) {
      const float mnew = fmaxf(mrow[r], sM[r + 8 * lh]);
      corr[r] = __expf(mrow[r] - mnew);
      mrow[r] = mnew;
      lrow[r] *= corr[r];
      float p = __expf(s[r] - mnew);
      float rs = p;
#pragma unroll
      for (int m = 1; m < 16; m <<= 1) rs += __shfl_xor(rs, m, 32);
      if (ln16 == 0) sRedS[w * 16 + r + 8 * lh] = rs;   // ungated partial sum
      p *= sg;                                          // post-softmax key gate
      sP[(r + 8 * lh) * 64 + w * 16 + ln16] = (__bf16)p;
    }
#pragma unroll
    for (int nt = 0; nt < 12; nt++)
#pragma unroll
      for (int r = 0; r < 8; r++) o[nt][r] *= corr[r];
    __syncthreads();
    if (tid < 16)
      sL[tid] = sRedS[tid] + sRedS[16 + tid] + sRedS[32 + tid] + sRedS[48 + tid];

    // ---- O += P @ V^T slice (wave columns w*192 .. +191) ----
#pragma unroll
    for (int ks2 = 0; ks2 < 2; ks2++) {
      const __bf16* pp = &sP[ln16 * 64 + ks2 * 32 + (lh ? 8 : 0)];
      v16bf a = ld_frag(pp, pp + 16);
#pragma unroll
      for (int nt = 0; nt < 12; nt++) {
        const int dcol = w * 192 + nt * 16 + ln16;
        const __bf16* vb = vt + ((size_t)bb * DD + dcol) * TT + kbase + ks2 * 32 + lh * 8;
        v16bf b = ld_frag(vb, vb + 16);
        o[nt] = WMMA_BF16(a, b, o[nt]);
      }
    }
    __syncthreads();
#pragma unroll
    for (int r = 0; r < 8; r++) lrow[r] += sL[r + 8 * lh];
  }

  // ---- finalize: scale by alpha[q] / l[q] and store f32 ----
#pragma unroll
  for (int r = 0; r < 8; r++) {
    const int row = q0 + r + 8 * lh;
    const float f = alpha[(size_t)bb * TT + row] / lrow[r];
    float* orow = out + ((size_t)bb * TT + row) * DD + w * 192;
#pragma unroll
    for (int nt = 0; nt < 12; nt++)
      orow[nt * 16 + ln16] = o[nt][r] * f;
  }
}

// ---------------------------------------------------------------------------
// Host-side orchestration (all on `stream`, scratch from d_ws):
//   ws layout: raw f32 [16384x768] (50.3 MB, reused; later aliased as V^T bf16)
//              q_ln bf16 (25.2 MB) | k_ln bf16 (25.2 MB) | alpha | sigma
// ---------------------------------------------------------------------------
extern "C" void kernel_launch(void* const* d_in, const int* in_sizes, int n_in,
                              void* d_out, int out_size, void* d_ws, size_t ws_size,
                              hipStream_t stream) {
  (void)in_sizes; (void)n_in; (void)out_size; (void)ws_size;
  const float* x    = (const float*)d_in[0];
  const float* Wq   = (const float*)d_in[1];
  const float* Wk   = (const float*)d_in[2];
  const float* Wv   = (const float*)d_in[3];
  const float* qn_g = (const float*)d_in[4];
  const float* qn_b = (const float*)d_in[5];
  const float* kn_g = (const float*)d_in[6];
  const float* kn_b = (const float*)d_in[7];
  const float* Wsig = (const float*)d_in[8];
  const float* bsig = (const float*)d_in[9];
  const float* Walp = (const float*)d_in[10];
  const float* balp = (const float*)d_in[11];
  float* out = (float*)d_out;

  char* ws = (char*)d_ws;
  const size_t RAW_BYTES = (size_t)BTOK * DD * sizeof(float);     // 50331648
  const size_t LN_BYTES  = (size_t)BTOK * DD * sizeof(__bf16);    // 25165824
  float*  raw   = (float*)ws;
  __bf16* q_ln  = (__bf16*)(ws + RAW_BYTES);
  __bf16* k_ln  = (__bf16*)(ws + RAW_BYTES + LN_BYTES);
  float*  alpha = (float*)(ws + RAW_BYTES + 2 * LN_BYTES);
  float*  sigma = alpha + BTOK;
  __bf16* vt    = (__bf16*)ws;   // aliases raw: safe, raw is dead by then

  const dim3 ggrid(BTOK / 128, DD / 128);   // 128 x 6

  // q = LN(x Wq) + gates
  gemm768_bf16<0><<<ggrid, 256, 0, stream>>>(x, Wq, raw);
  ln_gate<<<BTOK, 256, 0, stream>>>(raw, qn_g, qn_b, q_ln,
                                    Walp, balp, Wsig, bsig, alpha, sigma, 1);
  // k = LN(x Wk)
  gemm768_bf16<0><<<ggrid, 256, 0, stream>>>(x, Wk, raw);
  ln_gate<<<BTOK, 256, 0, stream>>>(raw, kn_g, kn_b, k_ln,
                                    Walp, balp, Wsig, bsig, alpha, sigma, 0);
  // v^T (bf16) into the raw buffer
  gemm768_bf16<1><<<ggrid, 256, 0, stream>>>(x, Wv, (void*)vt);

  // gated flash attention
  attn_flash<<<dim3(TT / 16, NBATCH), 128, 0, stream>>>(q_ln, k_ln, vt,
                                                        alpha, sigma, out);
}
